// SelfSpatialAttn_88184268521532
// MI455X (gfx1250) — compile-verified, matrix operand order checked by reading, not run on hardware
//
#include <hip/hip_runtime.h>
#include <hip/hip_bf16.h>

// Problem constants (from reference): N=16, C=128, H=128, W=128, fp32 in/out.
#define Nn 16
#define Cc 128
#define Hh 128
#define Ww 128
static constexpr int HW   = Hh * Ww;        // 16384
static constexpr int CHW  = Cc * HW;        // 2,097,152
static constexpr int LDK  = 136;            // bf16-unit stride for [row][K] LDS tiles (mult of 8 -> 16B rows)
static constexpr int LDS_SIM = 132;         // f32 stride for simT (mult of 4 for float4 scans)
static constexpr int NCHUNK = 4;            // images per chunk: kqv slab 48MB + x slice 32MB stay L2-resident
static constexpr size_t CHUNK_ELEMS = (size_t)NCHUNK * CHW;   // per-projection chunk elems

typedef float  v8f   __attribute__((ext_vector_type(8)));
typedef __bf16 v16bf __attribute__((ext_vector_type(16)));
typedef __bf16 v8bf  __attribute__((ext_vector_type(8)));
typedef __bf16 v4bf  __attribute__((ext_vector_type(4)));

// Per-lane A/B operand for V_WMMA_F32_16X16X32_BF16: two contiguous 8-bf16 runs
// (K = k0 + 8*half  and  K = k0 + 16 + 8*half) concatenated into one v16bf.
static __device__ inline v16bf ld16(const __bf16* run0, const __bf16* run1) {
    v8bf lo = *(const v8bf*)run0;
    v8bf hi = *(const v8bf*)run1;
    return __builtin_shufflevector(lo, hi, 0,1,2,3,4,5,6,7,8,9,10,11,12,13,14,15);
}

// ---------------------------------------------------------------------------
// Kernel 1: k/q/v projections, bf16 WMMA, one workgroup per (n,h) slice.
//   A = W_p[o][c] (bf16, natural row-major)   B = x^T[w][c] (bf16, LDS transpose)
//   kqv[p][n-n0][o][h][w] = bf16( sum_c W_p[o,c]*x[n,c,h,w] + b_p[o] )
// ---------------------------------------------------------------------------
__global__ void __launch_bounds__(256) proj_kernel(
    const float* __restrict__ x,
    const float* __restrict__ wk, const float* __restrict__ bk,
    const float* __restrict__ wq, const float* __restrict__ bq,
    const float* __restrict__ wv, const float* __restrict__ bv,
    __bf16* __restrict__ kqv, int n0)
{
    extern __shared__ char smem_raw[];
    __bf16* xT   = (__bf16*)smem_raw;            // [w][c] : 128*LDK bf16
    __bf16* wb   = xT + 128 * LDK;               // 3 * [o][c] : 128*LDK bf16 each
    float*  bias = (float*)(wb + 3 * 128 * LDK); // 3*128 f32

    const int t   = threadIdx.x;
    const int blk = blockIdx.x;
    const int nl  = blk >> 7;                    // chunk-local n
    const int n   = n0 + nl;
    const int h   = blk & 127;

    const float* wptr[3] = { wk, wq, wv };
    const float* bptr[3] = { bk, bq, bv };

    // ---- stage x[n,:,h,:] -> xT[w][c] (cvt + transpose) ----
    const float* xrow = x + (size_t)n * CHW + (size_t)h * Ww;
    for (int e = t; e < 128 * 32; e += 256) {          // float4 chunks
        const int c  = e >> 5;
        const int w4 = e & 31;
        float4 v = *(const float4*)(xrow + (size_t)c * HW + w4 * 4);
        xT[(w4 * 4 + 0) * LDK + c] = (__bf16)v.x;
        xT[(w4 * 4 + 1) * LDK + c] = (__bf16)v.y;
        xT[(w4 * 4 + 2) * LDK + c] = (__bf16)v.z;
        xT[(w4 * 4 + 3) * LDK + c] = (__bf16)v.w;
    }
    // ---- stage weights row-major bf16 ----
    for (int p = 0; p < 3; ++p) {
        const float* wp  = wptr[p];
        __bf16*      wbp = wb + p * 128 * LDK;
        for (int e = t; e < 128 * 128; e += 256) {
            const int o = e >> 7;
            const int c = e & 127;
            wbp[o * LDK + c] = (__bf16)wp[e];
        }
    }
    for (int e = t; e < 3 * 128; e += 256)
        bias[e] = bptr[e >> 7][e & 127];
    __syncthreads();

    const int wave = t >> 5;          // wave32
    const int lane = t & 31;
    const int lh   = lane >> 4;       // half-wave selects K-run offset
    const int l16  = lane & 15;

    for (int tile = wave; tile < 192; tile += 8) {
        const int p   = tile / 64;
        const int rem = tile - p * 64;
        const int tm  = rem >> 3;     // o-tile
        const int tn  = rem & 7;      // w-tile
        const __bf16* Arow = wb + p * 128 * LDK + (tm * 16 + l16) * LDK + 8 * lh;
        const __bf16* Brow = xT + (tn * 16 + l16) * LDK + 8 * lh;

        v8f acc = {};
        #pragma unroll
        for (int k0 = 0; k0 < 128; k0 += 32) {
            v16bf a = ld16(Arow + k0, Arow + k0 + 16);
            v16bf b = ld16(Brow + k0, Brow + k0 + 16);
            acc = __builtin_amdgcn_wmma_f32_16x16x32_bf16(
                      false, a, false, b, (short)0, acc, false, false);
        }
        __bf16* dst = kqv + (size_t)p * CHUNK_ELEMS + (size_t)nl * CHW + (size_t)h * Ww;
        #pragma unroll
        for (int i = 0; i < 8; ++i) {
            const int orow = tm * 16 + i + 8 * lh;
            dst[(size_t)orow * HW + tn * 16 + l16] =
                (__bf16)(acc[i] + bias[p * 128 + orow]);
        }
    }
}

// ---------------------------------------------------------------------------
// Kernel 2: fused attention per (n,o).  bf16 WMMA, fp32 accumulate/softmax.
//   simT[v][w] = (Q^T K)[v][w]/sqrt(C)  (A=qT rows, B=kT rows)
//   row softmax over w, exp -> attnT[v][w] bf16 (unnormalized, inv folded later)
//   out[h][v] = (V attn)[h][v] * g*inv[v] + x
// ---------------------------------------------------------------------------
__global__ void __launch_bounds__(256) attn_kernel(
    const float* __restrict__ x,
    const __bf16* __restrict__ kqv,
    const float* __restrict__ gamma,
    float* __restrict__ outp, int n0)
{
    extern __shared__ char smem_raw[];
    __bf16* kT   = (__bf16*)smem_raw;            // [w][h]
    __bf16* qT   = kT + 128 * LDK;               // [v][h]
    __bf16* vs   = qT + 128 * LDK;               // [h][w]
    __bf16* aT   = vs + 128 * LDK;               // attnT [v][w] (exp, bf16)
    float*  simT = (float*)(aT + 128 * LDK);     // [v][LDS_SIM] f32
    float*  red  = simT + 128 * LDS_SIM;         // 256
    float*  inv  = red + 256;                    // 128

    const int t   = threadIdx.x;
    const int blk = blockIdx.x;
    const int nl  = blk >> 7;                    // chunk-local n
    const int n   = n0 + nl;
    const int o   = blk & 127;

    const size_t cbase = (size_t)nl * CHW + (size_t)o * HW;   // chunk-local kqv base
    const size_t base  = (size_t)n  * CHW + (size_t)o * HW;   // absolute x/out base
    const __bf16* Kg = kqv + cbase;
    const __bf16* Qg = kqv + CHUNK_ELEMS + cbase;
    const __bf16* Vg = kqv + 2 * CHUNK_ELEMS + cbase;

    // ---- stage: K,Q transposed to [w/v][h]; V row-major [h][w] ----
    for (int e = t; e < 128 * 16; e += 256) {         // 8-bf16 chunks
        const int h  = e >> 4;
        const int w8 = e & 15;
        v8bf kv = *(const v8bf*)(Kg + h * Ww + w8 * 8);
        v8bf qv = *(const v8bf*)(Qg + h * Ww + w8 * 8);
        #pragma unroll
        for (int j = 0; j < 8; ++j) {
            kT[(w8 * 8 + j) * LDK + h] = kv[j];
            qT[(w8 * 8 + j) * LDK + h] = qv[j];
        }
        *(v8bf*)(vs + h * LDK + w8 * 8) = *(const v8bf*)(Vg + h * Ww + w8 * 8);
    }
    __syncthreads();

    const int wave = t >> 5;
    const int lane = t & 31;
    const int lh   = lane >> 4;
    const int l16  = lane & 15;
    const float rscale = 0.088388347648318447f;       // 1/sqrt(128)

    // ---- simT = Q^T K * rscale ----
    for (int tile = wave; tile < 64; tile += 8) {
        const int tm = tile >> 3, tn = tile & 7;      // tm: v-tile, tn: w-tile
        const __bf16* Arow = qT + (tm * 16 + l16) * LDK + 8 * lh;
        const __bf16* Brow = kT + (tn * 16 + l16) * LDK + 8 * lh;
        v8f acc = {};
        #pragma unroll
        for (int k0 = 0; k0 < 128; k0 += 32) {
            v16bf a = ld16(Arow + k0, Arow + k0 + 16);
            v16bf b = ld16(Brow + k0, Brow + k0 + 16);
            acc = __builtin_amdgcn_wmma_f32_16x16x32_bf16(
                      false, a, false, b, (short)0, acc, false, false);
        }
        #pragma unroll
        for (int i = 0; i < 8; ++i)
            simT[(tm * 16 + i + 8 * lh) * LDS_SIM + tn * 16 + l16] = acc[i] * rscale;
    }
    __syncthreads();

    // ---- softmax over w per row v (2 threads/row, float4 scans) ----
    {
        const int v  = t & 127;
        const int w0 = (t >> 7) * 64;
        const float* srow = simT + v * LDS_SIM;
        float m = -3.402823466e38f;
        for (int w = w0; w < w0 + 64; w += 4) {
            float4 s4 = *(const float4*)(srow + w);
            m = fmaxf(m, fmaxf(fmaxf(s4.x, s4.y), fmaxf(s4.z, s4.w)));
        }
        red[t] = m;
        __syncthreads();
        const float mm = fmaxf(red[v], red[v + 128]);
        __syncthreads();
        float s = 0.0f;
        for (int w = w0; w < w0 + 64; w += 4) {
            float4 s4 = *(const float4*)(srow + w);
            float e0 = __expf(s4.x - mm), e1 = __expf(s4.y - mm);
            float e2 = __expf(s4.z - mm), e3 = __expf(s4.w - mm);
            v4bf ev = { (__bf16)e0, (__bf16)e1, (__bf16)e2, (__bf16)e3 };
            *(v4bf*)(aT + v * LDK + w) = ev;      // 8B store, 8B aligned
            s += (e0 + e1) + (e2 + e3);
        }
        red[t] = s;
        __syncthreads();
        if (t < 128) inv[v] = 1.0f / (red[v] + red[v + 128]);
    }
    __syncthreads();

    const float g = gamma[0];

    // ---- out[h][v] = (V @ exp_attn)[h][v] * g*inv[v] + x ----
    for (int tile = wave; tile < 64; tile += 8) {
        const int tm = tile >> 3, tn = tile & 7;      // tm: h-tile, tn: v-tile
        const __bf16* Arow = vs + (tm * 16 + l16) * LDK + 8 * lh;
        const __bf16* Brow = aT + (tn * 16 + l16) * LDK + 8 * lh;
        v8f acc = {};
        #pragma unroll
        for (int k0 = 0; k0 < 128; k0 += 32) {
            v16bf a = ld16(Arow + k0, Arow + k0 + 16);
            v16bf b = ld16(Brow + k0, Brow + k0 + 16);
            acc = __builtin_amdgcn_wmma_f32_16x16x32_bf16(
                      false, a, false, b, (short)0, acc, false, false);
        }
        const float sc = g * inv[tn * 16 + l16];
        const float* xb = x + base;
        float*       ob = outp + base;
        #pragma unroll
        for (int i = 0; i < 8; ++i) {
            const int hrow = tm * 16 + i + 8 * lh;
            const int idx  = hrow * Ww + tn * 16 + l16;
            ob[idx] = acc[i] * sc + xb[idx];
        }
    }
}

// ---------------------------------------------------------------------------
extern "C" void kernel_launch(void* const* d_in, const int* in_sizes, int n_in,
                              void* d_out, int out_size, void* d_ws, size_t ws_size,
                              hipStream_t stream) {
    const float* x     = (const float*)d_in[0];
    const float* wk    = (const float*)d_in[1];
    const float* bk    = (const float*)d_in[2];
    const float* wq    = (const float*)d_in[3];
    const float* bq    = (const float*)d_in[4];
    const float* wv    = (const float*)d_in[5];
    const float* bv    = (const float*)d_in[6];
    const float* gamma = (const float*)d_in[7];
    float*  out = (float*)d_out;
    __bf16* kqv = (__bf16*)d_ws;   // 3 * NCHUNK*C*H*W bf16 = 48 MiB, reused per chunk (L2-resident)

    const size_t smemA = (size_t)(4 * 128 * LDK) * sizeof(__bf16)
                       + (size_t)(3 * 128) * sizeof(float);
    const size_t smemB = (size_t)(4 * 128 * LDK) * sizeof(__bf16)
                       + (size_t)(128 * LDS_SIM + 256 + 128) * sizeof(float);

    // Chunk the batch so the kqv slab (48MB) + x slice (32MB) stay resident in the
    // 192MB global L2 between the projection pass and the attention pass: the
    // bf16 staging never round-trips HBM.  proj/attn alternate on `stream`, so
    // ordering is guaranteed and the workspace slab is safely reused per chunk.
    for (int n0 = 0; n0 < Nn; n0 += NCHUNK) {
        proj_kernel<<<NCHUNK * Hh, 256, smemA, stream>>>(x, wk, bk, wq, bq, wv, bv, kqv, n0);
        attn_kernel<<<NCHUNK * Cc, 256, smemB, stream>>>(x, kqv, gamma, out, n0);
    }
}
